// MutualCrossAttention_80367428042874
// MI455X (gfx1250) — matrix operand order repaired and checked by reading, not compiled, not run on hardware
//
#include <hip/hip_runtime.h>

// Problem constants (from reference): B=8, L=2048, D=512, fp32 in/out.
#define BB    8
#define LL    2048
#define DD    512
#define MTILE 32            // query rows per block (2 x 16-row WMMA tiles)
#define KBLK  64            // keys per iteration (two WMMA K=32 steps for P*V)
#define KROW  260           // dwords per K LDS row: 256 data + 4 pad (bank-conflict free)
#define VROW  36            // dwords per Vt/P LDS row: 32 data + 4 pad

typedef __attribute__((ext_vector_type(16))) __bf16 v16bf;
typedef __attribute__((ext_vector_type(8)))  float  v8f;

union BfVec {
    uint4 q[2];
    v16bf v;
};

__device__ __forceinline__ unsigned short f2bf(float f) {
    unsigned u = __builtin_bit_cast(unsigned, f);
    unsigned r = ((u >> 16) & 1u) + 0x7FFFu;   // round-to-nearest-even
    return (unsigned short)((u + r) >> 16);
}
__device__ __forceinline__ unsigned pack2(float a, float b) {
    return (unsigned)f2bf(a) | ((unsigned)f2bf(b) << 16);
}

#define WMMA_BF16(A, Bv, C) \
    __builtin_amdgcn_wmma_f32_16x16x32_bf16(false, (A), false, (Bv), (short)0, (C), false, false)

// ---------------- one-time fp32 -> bf16 pre-convert kernels (fast path) ----------------

__global__ __launch_bounds__(256)
void cvt_bf16_kernel(const float* __restrict__ src, unsigned short* __restrict__ dst, int n8)
{
    int i = blockIdx.x * blockDim.x + threadIdx.x;
    if (i >= n8) return;
    const float4* s = (const float4*)src;
    float4 f0 = s[2 * i];
    float4 f1 = s[2 * i + 1];
    uint4 w;
    w.x = pack2(f0.x, f0.y);
    w.y = pack2(f0.z, f0.w);
    w.z = pack2(f1.x, f1.y);
    w.w = pack2(f1.z, f1.w);
    ((uint4*)dst)[i] = w;
}

// dst[b][d][l] = bf16(src[b][l][d]), 32x32 tiles through LDS.
__global__ __launch_bounds__(256)
void cvt_tr_kernel(const float* __restrict__ src, unsigned short* __restrict__ dst)
{
    __shared__ float tile[32][33];
    const int tilesPerRow   = DD / 32;                 // 16
    const int tilesPerBatch = (LL / 32) * tilesPerRow; // 1024
    const int b  = blockIdx.x / tilesPerBatch;
    const int r  = blockIdx.x % tilesPerBatch;
    const int l0 = (r / tilesPerRow) * 32;
    const int d0 = (r % tilesPerRow) * 32;
    const int t  = threadIdx.x;
    const int lr = t >> 3;         // 0..31
    const int lc = (t & 7) * 4;    // 0..28

    float4 f = *(const float4*)(src + ((size_t)b * LL + l0 + lr) * DD + d0 + lc);
    tile[lr][lc + 0] = f.x;
    tile[lr][lc + 1] = f.y;
    tile[lr][lc + 2] = f.z;
    tile[lr][lc + 3] = f.w;
    __syncthreads();

    uint2 w;
    w.x = pack2(tile[lc + 0][lr], tile[lc + 1][lr]);
    w.y = pack2(tile[lc + 2][lr], tile[lc + 3][lr]);
    *(uint2*)(dst + ((size_t)b * DD + d0 + lr) * LL + l0 + lc) = w;
}

// ---------------- flash attention pass (K == V) ----------------
// out[b,q,:] (+)= softmax(Q K^T * scale) V.
// Waves: mr = wave&1 (16-row query tile), dh = wave>>1 (128-wide D slice AND
// 128-wide slice of the K=512 reduction of S). Partial S tiles are summed with
// ds_add_f32 into one LDS buffer so every WMMA issued is unique work.
template <bool PRECONV>
__global__ __launch_bounds__(256)
void mca_flash_attn(const float* __restrict__ xq32, const float* __restrict__ xk32,
                    const unsigned short* __restrict__ xqb,
                    const unsigned short* __restrict__ xkb,
                    const unsigned short* __restrict__ xkT,   // [b][d][l] bf16
                    float* __restrict__ out, int accumulate)
{
    extern __shared__ __align__(16) unsigned smem[];
    unsigned* Ks  = smem;                       // KBLK x KROW  row-major keys (bf16)
    unsigned* Vts = Ks  + KBLK * KROW;          // DD   x VROW  d-major keys (bf16)
    unsigned* Ps  = Vts + DD   * VROW;          // 2 x 16 x VROW P matrices (bf16)
    float*    Sb  = (float*)(Ps + 2 * 16 * VROW); // 2mr x 4tile x 8v x 32lane summed S
    unsigned* SbU = (unsigned*)Sb;              // 2048 dwords

    const int tid  = threadIdx.x;
    const int wave = tid >> 5;
    const int lane = tid & 31;
    const int lan  = lane & 15;
    const int h    = lane >> 4;
    const int mr   = wave & 1;      // query row tile
    const int dh   = wave >> 1;     // 128-wide D slice (0..3)

    const int qblocks = LL / MTILE;             // 64
    const int b     = blockIdx.x / qblocks;
    const int qbase = (blockIdx.x % qblocks) * MTILE;

    // logits handled in log2 domain: raw_dot * (1/sqrt(512)) * log2(e)
    const float c2 = 0.06372648297098931f;

    // ---- preload this wave's loop-invariant Q A-operands (4 chunks) into registers ----
    BfVec qreg[4];
    if (PRECONV) {
        const unsigned short* qr = xqb + ((size_t)b * LL + qbase + mr * 16 + lan) * DD;
#pragma unroll
        for (int cc = 0; cc < 4; ++cc) {
            const int c = dh * 4 + cc;
            qreg[cc].q[0] = *(const uint4*)(qr + 32 * c + 8 * h);
            qreg[cc].q[1] = *(const uint4*)(qr + 32 * c + 16 + 8 * h);
        }
    } else {
        const float* qr = xq32 + ((size_t)b * LL + qbase + mr * 16 + lan) * DD;
#pragma unroll
        for (int cc = 0; cc < 4; ++cc) {
            const int c = dh * 4 + cc;
            float4 g0 = *(const float4*)(qr + 32 * c + 8 * h);
            float4 g1 = *(const float4*)(qr + 32 * c + 8 * h + 4);
            float4 g2 = *(const float4*)(qr + 32 * c + 16 + 8 * h);
            float4 g3 = *(const float4*)(qr + 32 * c + 16 + 8 * h + 4);
            uint4 w;
            w.x = pack2(g0.x, g0.y); w.y = pack2(g0.z, g0.w);
            w.z = pack2(g1.x, g1.y); w.w = pack2(g1.z, g1.w);
            qreg[cc].q[0] = w;
            w.x = pack2(g2.x, g2.y); w.y = pack2(g2.z, g2.w);
            w.z = pack2(g3.x, g3.y); w.w = pack2(g3.z, g3.w);
            qreg[cc].q[1] = w;
        }
    }

    // O accumulators: 8 tiles -> rows v+8h, cols dh*128 + t*16 + lan
    v8f o[8];
#pragma unroll
    for (int t = 0; t < 8; ++t)
#pragma unroll
        for (int v = 0; v < 8; ++v) o[t][v] = 0.0f;

    float mst[8], lst[8];
#pragma unroll
    for (int v = 0; v < 8; ++v) { mst[v] = -3.0e38f; lst[v] = 0.0f; }

    const int nkb = LL / KBLK;                   // 32
#pragma unroll 1
    for (int kb = 0; kb < nkb; ++kb) {
        __syncthreads();                         // prev iter's Ks/Vts/Ps/Sb reads done

        // ---- zero the S reduction buffer (2048 dwords) ----
        {
            uint4 z = {0u, 0u, 0u, 0u};
            *(uint4*)(SbU + tid * 8)     = z;
            *(uint4*)(SbU + tid * 8 + 4) = z;
        }

        // ---- stage K/V tile: row-major Ks and d-major Vts ----
        if (PRECONV) {
            {
                const int row = tid >> 2;        // key 0..63
                const int c0  = (tid & 3) * 64;  // dwords
                const unsigned short* g = xkb + ((size_t)b * LL + (size_t)kb * KBLK + row) * DD + c0 * 2;
                const uint4* src = (const uint4*)g;
                unsigned* dst = Ks + row * KROW + c0;
#pragma unroll
                for (int i = 0; i < 16; ++i) *(uint4*)(dst + 4 * i) = src[i];
                if (kb + 1 < nkb)
                    __builtin_prefetch(g + (size_t)KBLK * DD, 0, 1);   // global_prefetch_b8
            }
            {
                const int d0 = tid * 2;          // two Vt rows per thread
#pragma unroll
                for (int j = 0; j < 2; ++j) {
                    const int d = d0 + j;
                    const unsigned short* g = xkT + ((size_t)b * DD + d) * LL + (size_t)kb * KBLK;
                    const uint4* src = (const uint4*)g;
                    unsigned* dst = Vts + d * VROW;
#pragma unroll
                    for (int i = 0; i < 8; ++i) *(uint4*)(dst + 4 * i) = src[i];
                    if (kb + 1 < nkb)
                        __builtin_prefetch(g + KBLK, 0, 1);
                }
            }
        } else {
            const int row = tid >> 2;            // key 0..63
            const int d0  = (tid & 3) * 128;
            const float* srow = xk32 + ((size_t)b * LL + (size_t)kb * KBLK + row) * DD + d0;
            const float4* src = (const float4*)srow;
            unsigned* dst = Ks + row * KROW + (d0 >> 1);
            unsigned short* vt = (unsigned short*)Vts;
#pragma unroll
            for (int i = 0; i < 16; ++i) {
                float4 f0 = src[2 * i], f1 = src[2 * i + 1];
                uint4 w;
                w.x = pack2(f0.x, f0.y); w.y = pack2(f0.z, f0.w);
                w.z = pack2(f1.x, f1.y); w.w = pack2(f1.z, f1.w);
                *(uint4*)(dst + 4 * i) = w;
                const int d = d0 + 8 * i;
                vt[(size_t)(d + 0) * (2 * VROW) + row] = f2bf(f0.x);
                vt[(size_t)(d + 1) * (2 * VROW) + row] = f2bf(f0.y);
                vt[(size_t)(d + 2) * (2 * VROW) + row] = f2bf(f0.z);
                vt[(size_t)(d + 3) * (2 * VROW) + row] = f2bf(f0.w);
                vt[(size_t)(d + 4) * (2 * VROW) + row] = f2bf(f1.x);
                vt[(size_t)(d + 5) * (2 * VROW) + row] = f2bf(f1.y);
                vt[(size_t)(d + 6) * (2 * VROW) + row] = f2bf(f1.z);
                vt[(size_t)(d + 7) * (2 * VROW) + row] = f2bf(f1.w);
            }
            if (kb + 1 < nkb)
                __builtin_prefetch(srow + (size_t)KBLK * DD, 0, 1);
        }
        __syncthreads();

        // ---- partial S over this wave's 4 chunks (d = dh*128 .. dh*128+127) ----
        // 4 tiles of 16 keys each: s[j] covers keys 16j..16j+15 of the block.
        v8f s[4];
#pragma unroll
        for (int j = 0; j < 4; ++j)
#pragma unroll
            for (int v = 0; v < 8; ++v) s[j][v] = 0.0f;
        {
#pragma unroll
            for (int cc = 0; cc < 4; ++cc) {
                const int c = dh * 4 + cc;
                const unsigned* vrow = Ks + lan * KROW + 16 * c + 4 * h;
#pragma unroll
                for (int j = 0; j < 4; ++j) {
                    BfVec bv;
                    bv.q[0] = *(const uint4*)(vrow + 16 * j * KROW);
                    bv.q[1] = *(const uint4*)(vrow + 16 * j * KROW + 8);
                    s[j] = WMMA_BF16(qreg[cc].v, bv.v, s[j]);
                }
            }
        }
        // accumulate partials across the 4 dh waves (conflict-free ds_add_f32)
#pragma unroll
        for (int j = 0; j < 4; ++j)
#pragma unroll
            for (int v = 0; v < 8; ++v)
                atomicAdd(&Sb[((mr * 4 + j) * 8 + v) * 32 + lane], s[j][v]);
        __syncthreads();

        // ---- online softmax over this 64-key block (log2 domain) ----
        float p[4][8], alpha[8];
#pragma unroll
        for (int v = 0; v < 8; ++v) {
            float a0 = Sb[((mr * 4 + 0) * 8 + v) * 32 + lane] * c2;
            float a1 = Sb[((mr * 4 + 1) * 8 + v) * 32 + lane] * c2;
            float a2 = Sb[((mr * 4 + 2) * 8 + v) * 32 + lane] * c2;
            float a3 = Sb[((mr * 4 + 3) * 8 + v) * 32 + lane] * c2;
            float rm = fmaxf(fmaxf(a0, a1), fmaxf(a2, a3));
#pragma unroll
            for (int m = 1; m <= 8; m <<= 1)     // reduce over 16 lanes of each half
                rm = fmaxf(rm, __shfl_xor(rm, m, 32));
            float mn = fmaxf(mst[v], rm);
            alpha[v] = exp2f(mst[v] - mn);
            p[0][v] = exp2f(a0 - mn);
            p[1][v] = exp2f(a1 - mn);
            p[2][v] = exp2f(a2 - mn);
            p[3][v] = exp2f(a3 - mn);
            float rs = (p[0][v] + p[1][v]) + (p[2][v] + p[3][v]);
#pragma unroll
            for (int m = 1; m <= 8; m <<= 1)
                rs += __shfl_xor(rs, m, 32);
            lst[v] = lst[v] * alpha[v] + rs;
            mst[v] = mn;
        }
#pragma unroll
        for (int t = 0; t < 8; ++t)
#pragma unroll
            for (int v = 0; v < 8; ++v) o[t][v] *= alpha[v];

        // ---- write P (bf16, row-major 16x64); identical across dh -> dh==0 writes ----
        if (dh == 0) {
            unsigned short* P = (unsigned short*)(Ps + mr * 16 * VROW);
#pragma unroll
            for (int v = 0; v < 8; ++v) {
                const int row = v + 8 * h;
#pragma unroll
                for (int j = 0; j < 4; ++j)
                    P[row * (2 * VROW) + lan + 16 * j] = f2bf(p[j][v]);
            }
        }
        __syncthreads();

        // ---- O += P * V (K = 64 keys = two K=32 WMMA steps per 16-col tile) ----
        {
            BfVec a0, a1;
            const unsigned* prow = Ps + (mr * 16 + lan) * VROW;
            a0.q[0] = *(const uint4*)(prow + 4 * h);
            a0.q[1] = *(const uint4*)(prow + 4 * h + 8);
            a1.q[0] = *(const uint4*)(prow + 16 + 4 * h);
            a1.q[1] = *(const uint4*)(prow + 16 + 4 * h + 8);
            const unsigned* k0 = Vts + (dh * 128 + lan) * VROW + 4 * h;        // keys 0..31
            const unsigned* k1 = Vts + (dh * 128 + lan) * VROW + 16 + 4 * h;   // keys 32..63
#pragma unroll
            for (int t = 0; t < 8; ++t) {
                const int off = t * 16 * VROW;
                BfVec bv;
                bv.q[0] = *(const uint4*)(k0 + off);
                bv.q[1] = *(const uint4*)(k0 + off + 8);
                o[t] = WMMA_BF16(a0.v, bv.v, o[t]);
                bv.q[0] = *(const uint4*)(k1 + off);
                bv.q[1] = *(const uint4*)(k1 + off + 8);
                o[t] = WMMA_BF16(a1.v, bv.v, o[t]);
            }
        }
    }

    // ---- normalize and store / accumulate ----
    float inv[8];
#pragma unroll
    for (int v = 0; v < 8; ++v) inv[v] = 1.0f / lst[v];

#pragma unroll
    for (int t = 0; t < 8; ++t) {
        const int col = dh * 128 + t * 16 + lan;
#pragma unroll
        for (int v = 0; v < 8; ++v) {
            const int row = qbase + mr * 16 + v + 8 * h;
            const size_t idx = ((size_t)b * LL + row) * DD + col;
            float val = o[t][v] * inv[v];
            if (accumulate) val += out[idx];
            out[idx] = val;
        }
    }
}

extern "C" void kernel_launch(void* const* d_in, const int* in_sizes, int n_in,
                              void* d_out, int out_size, void* d_ws, size_t ws_size,
                              hipStream_t stream) {
    (void)in_sizes; (void)n_in; (void)out_size;
    const float* x1 = (const float*)d_in[0];
    const float* x2 = (const float*)d_in[1];
    float* out = (float*)d_out;

    const size_t nElem    = (size_t)BB * LL * DD;        // 8,388,608
    const size_t bfBytes  = nElem * sizeof(unsigned short);
    const size_t wsNeeded = 4 * bfBytes;                 // x1b, x2b, x1T, x2T = 64 MB

    const size_t smem_bytes =
        (size_t)(KBLK * KROW + DD * VROW + 2 * 16 * VROW + 2 * 4 * 8 * 32) * sizeof(unsigned);

    dim3 grid(BB * (LL / MTILE));   // 512 blocks
    dim3 block(256);

    if (d_ws != nullptr && ws_size >= wsNeeded) {
        unsigned short* x1b = (unsigned short*)d_ws;
        unsigned short* x2b = x1b + nElem;
        unsigned short* x1T = x2b + nElem;
        unsigned short* x2T = x1T + nElem;

        const int n8 = (int)(nElem / 8);
        dim3 cg((n8 + 255) / 256);
        cvt_bf16_kernel<<<cg, block, 0, stream>>>(x1, x1b, n8);
        cvt_bf16_kernel<<<cg, block, 0, stream>>>(x2, x2b, n8);
        dim3 tg(BB * (LL / 32) * (DD / 32));    // 8192 tiles
        cvt_tr_kernel<<<tg, block, 0, stream>>>(x1, x1T);
        cvt_tr_kernel<<<tg, block, 0, stream>>>(x2, x2T);

        // Pass A: out  = attn(Q=x1, K=V=x2)
        mca_flash_attn<true><<<grid, block, smem_bytes, stream>>>(x1, x2, x1b, x2b, x2T, out, 0);
        // Pass B: out += attn(Q=x2, K=V=x1)
        mca_flash_attn<true><<<grid, block, smem_bytes, stream>>>(x2, x1, x2b, x1b, x1T, out, 1);
    } else {
        mca_flash_attn<false><<<grid, block, smem_bytes, stream>>>(x1, x2, nullptr, nullptr, nullptr, out, 0);
        mca_flash_attn<false><<<grid, block, smem_bytes, stream>>>(x2, x1, nullptr, nullptr, nullptr, out, 1);
    }
}
